// OSFCwSTU_15238543966869
// MI455X (gfx1250) — matrix operand order repaired and checked by reading, not compiled, not run on hardware
//
#include <hip/hip_runtime.h>
#include <hip/hip_bf16.h>

// ---------------------------------------------------------------------------
// OSFC-with-STU, MI455X (gfx1250).
// Sequential 512-step recurrence -> single persistent workgroup (32 wave32
// waves on one WGP). Rolling histories live in LDS (320KB/WGP). The circular-
// buffer rotation is materialized ONCE per step into rotated phi^T copies in
// LDS, so both WMMA operands load as contiguous 128-bit fragments
// (global_load_b128 for W rows, ds_load_b128 for phi/U rows).
// bf16 WMMA (v_wmma_f32_16x16x32_bf16) for the two history matmuls; the
// Adagrad update of M is fused into the next step's <M,v> contraction so
// M/G stream through L2 exactly once per step (~4MB/step, L2-resident).
// ---------------------------------------------------------------------------

typedef __bf16 bf16;
typedef __bf16 v16bf __attribute__((ext_vector_type(16)));
typedef float  v8f   __attribute__((ext_vector_type(8)));
typedef unsigned int u32x4 __attribute__((ext_vector_type(4)));

#define DD   256
#define NN   64
#define HH   16
#define TT   512
#define HSTU 20
#define ETA  1e-3f
#define EPS  1e-10f

// A-fragment (16x32 bf16, MxK): lane L holds row M=L&15; half=L>>4 selects
// K subsets {0..7,16..23} (half 0) / {8..15,24..31} (half 1). Row-major
// memory gives two contiguous 16B loads per lane.
__device__ __forceinline__ v16bf load_A(const bf16* rowbase, int kc, int half) {
  union { v16bf v; u32x4 q[2]; } a;
  const bf16* p = rowbase + kc + half * 8;
  a.q[0] = *(const u32x4*)(p);
  a.q[1] = *(const u32x4*)(p + 16);
  return a.v;
}

// B-fragment (32x16 bf16, KxN): lane L is column N=L&15; half=L>>4 selects
// K subsets {0..7,16..23} (half 0) / {8..15,24..31} (half 1), matching A. rowbase
// points at the per-step ROTATED phi^T row N, so no index math remains here.
__device__ __forceinline__ v16bf load_B(const bf16* rowbase, int kc, int half) {
  union { v16bf v; u32x4 q[2]; } b;
  const bf16* p = rowbase + kc + half * 8;
  b.q[0] = *(const u32x4*)(p);
  b.q[1] = *(const u32x4*)(p + 16);
  return b.v;
}

__global__ __launch_bounds__(1024, 1) void osfc_stu_kernel(
    const float* __restrict__ A, const float* __restrict__ Bm,
    const float* __restrict__ Q, const float* __restrict__ R,
    const float* __restrict__ M0, const float* __restrict__ Mstu,
    const float* __restrict__ x0, const float* __restrict__ w0,
    const float* __restrict__ phi, const float* __restrict__ phistu,
    float* __restrict__ out,
    float* __restrict__ Mw, float* __restrict__ Gw, bf16* __restrict__ Wg) {
  // ---- LDS state (~215KB; legal on gfx1250: 320KB per WGP) ----
  __shared__ bf16  U_s[NN * TT];       // u history, circular, row-major [n][t]   64KB
  __shared__ bf16  phiT[HH * TT];      // phi^T  [h][t] bf16 (static)             16KB
  __shared__ bf16  phisT[32 * TT];     // phi_stu^T padded to 32 rows (static)    32KB
  __shared__ bf16  phiRot[HH * TT];    // per-step rotated phi^T                  16KB
  __shared__ bf16  phisRot[32 * TT];   // per-step rotated phi_stu^T              32KB
  __shared__ float vT[2][HH * DD];     // v^T ping-pong [h][d]                    32KB
  __shared__ float vu_s[NN * 32];      // U@phi_stu, cols 20..31 junk              8KB
  __shared__ float x_s[DD];
  __shared__ float u_s[NN];
  __shared__ float ru_s[2][NN];        // (R+R^T)u ping-pong
  __shared__ float red[1024];
  __shared__ float p1s[1024];
  __shared__ float p2s[1024];

  const int tid  = threadIdx.x;
  const int lane = tid & 31;
  const int wave = tid >> 5;

  // ------------------------- init (every launch) ---------------------------
  for (int i = tid; i < DD * TT; i += 1024) Wg[i] = (bf16)0.f;
  for (int i = tid; i < HH * NN * DD; i += 1024) { Mw[i] = M0[i]; Gw[i] = 0.f; }
  for (int i = tid; i < NN * TT; i += 1024) U_s[i] = (bf16)0.f;
  for (int i = tid; i < HH * TT; i += 1024) {
    int h = i >> 9, t = i & 511;
    phiT[i] = (bf16)phi[t * HH + h];
  }
  for (int i = tid; i < 32 * TT; i += 1024) {
    int h = i >> 9, t = i & 511;
    phisT[i] = (h < HSTU) ? (bf16)phistu[t * HSTU + h] : (bf16)0.f;
  }
  {
    float* vflat = &vT[0][0];
    for (int i = tid; i < 2 * HH * DD; i += 1024) vflat[i] = 0.f;
  }
  if (tid < DD) x_s[tid] = x0[tid];
  if (tid < NN) { u_s[tid] = 0.f; ru_s[0][tid] = 0.f; ru_s[1][tid] = 0.f; }
  __syncthreads();
  if (tid < DD) Wg[(size_t)tid * TT + 0] = (bf16)w0[tid];  // logical col 0 = w0
  __threadfence();
  __syncthreads();

  int head = 0;  // logical col j of W at physical (head + j) & 511
  for (int step = 0; step < TT; ++step) {
    const int p = step & 1;
    const int headu = (head - 1) & 511;  // head after this step's roll

    // -------- Phase 0: materialize rotated phi^T copies, zero u ------------
    if (tid < NN) u_s[tid] = 0.f;
    for (int i = tid; i < HH * TT; i += 1024) {          // 8 elems/thread
      const int n = i >> 9, c = i & 511;
      phiRot[i] = phiT[(n << 9) | ((c - head) & 511)];
    }
    for (int i = tid; i < 32 * TT; i += 1024) {          // 16 elems/thread
      const int n = i >> 9, c = i & 511;
      phisRot[i] = phisT[(n << 9) | ((c - headu) & 511)];
    }
    __syncthreads();

    // -------- Phase 1: v = W @ phi  (waves 0..15, bf16 WMMA, K=512) --------
    if (wave < 16) {
      const int half = lane >> 4, nl = lane & 15;
      const bf16* wrow = Wg + (size_t)(wave * 16 + nl) * TT;
      const bf16* brow = phiRot + nl * TT;
      v8f acc = {0.f, 0.f, 0.f, 0.f, 0.f, 0.f, 0.f, 0.f};
#pragma unroll 4
      for (int kc = 0; kc < TT; kc += 32) {
        __builtin_prefetch(wrow + kc + 128, 0, 3);  // global_prefetch_b8
        v16bf a = load_A(wrow, kc, half);
        v16bf b = load_B(brow, kc, half);
        acc = __builtin_amdgcn_wmma_f32_16x16x32_bf16(
            false, a, false, b, (short)0, acc, false, false);
      }
#pragma unroll
      for (int j = 0; j < 8; ++j)  // C layout: M = j + 8*half, N = nl
        vT[p][nl * DD + wave * 16 + j + 8 * half] = acc[j];
    }
    __syncthreads();

    // -------- Phase 2: fused Adagrad(M) from step-1 grads + u = <M, v> -----
    {
      const float* vprev = vT[p ^ 1];
      const float* vcur  = vT[p];
      const float* rup   = ru_s[p ^ 1];
      for (int pp = 0; pp < 32; ++pp) {
        const int pair = wave * 32 + pp;     // (h,n) pair; 32 waves x 32 = 1024
        const int h = pair >> 6, n = pair & 63;
        const int base = (h * NN + n) * DD;
        const float g0 = rup[n];
        float part = 0.f;
        for (int d0 = 0; d0 < DD; d0 += 32) {
          const int d = d0 + lane;
          const float g  = g0 * vprev[h * DD + d];     // gM = (Rsym u)[n] v[d,h]
          const float Gn = Gw[base + d] + g * g;
          Gw[base + d] = Gn;
          const float m = Mw[base + d] - ETA * g / (sqrtf(Gn) + EPS);
          Mw[base + d] = m;
          part += m * vcur[h * DD + d];
        }
        for (int off = 16; off > 0; off >>= 1) part += __shfl_xor(part, off, 32);
        if (lane == 0) atomicAdd(&u_s[n], part);       // ds_add_f32
      }
    }
    __syncthreads();

    // -------- Phase 3: cost = xQx + uRu, Rsym@u, insert u into U -----------
    {
      const int d = tid >> 2, q = tid & 3;
      float cp = 0.f;
      const float  xd = x_s[d];
      const float* Qr = Q + d * DD + q * 64;
      const float* xr = x_s + q * 64;
#pragma unroll 8
      for (int j = 0; j < 64; ++j) cp += xd * Qr[j] * xr[j];
#pragma unroll
      for (int e = 0; e < 4; ++e) {  // u^T R u over 4096 terms, 4 per thread
        const int idx = tid * 4 + e;
        const int n = idx >> 6, mc = idx & 63;
        cp += u_s[n] * R[n * 64 + mc] * u_s[mc];
      }
      red[tid] = cp;
      if (tid < NN) {
        float s = 0.f;
        const float* Rr = R + tid * 64;
        for (int mc = 0; mc < NN; ++mc) s += (Rr[mc] + R[mc * 64 + tid]) * u_s[mc];
        ru_s[p][tid] = s;
        U_s[tid * TT + headu] = (bf16)u_s[tid];  // roll + insert (circular)
      }
      __syncthreads();
      for (int s = 512; s > 0; s >>= 1) {
        if (tid < s) red[tid] += red[tid + s];
        __syncthreads();
      }
      if (tid == 0) out[step] = red[0];
    }

    // -------- Phase 4: vu = U @ phi_stu (waves 0..7, bf16 WMMA) ------------
    if (wave < 8) {
      const int rt = wave >> 1, ct = wave & 1;   // 4 n-tiles x 2 i-tiles
      const int half = lane >> 4, nl = lane & 15;
      const bf16* arow = U_s + (rt * 16 + nl) * TT;
      const bf16* brow = phisRot + (ct * 16 + nl) * TT;
      v8f acc = {0.f, 0.f, 0.f, 0.f, 0.f, 0.f, 0.f, 0.f};
#pragma unroll 4
      for (int kc = 0; kc < TT; kc += 32) {
        v16bf a = load_A(arow, kc, half);
        v16bf b = load_B(brow, kc, half);
        acc = __builtin_amdgcn_wmma_f32_16x16x32_bf16(
            false, a, false, b, (short)0, acc, false, false);
      }
#pragma unroll
      for (int j = 0; j < 8; ++j)
        vu_s[(rt * 16 + j + 8 * half) * 32 + ct * 16 + nl] = acc[j];
    }
    __syncthreads();

    // -------- Phase 5: x_next = <M_stu, vu>, new_w = x_next - Ax - Bu ------
    {
      const int d = tid >> 2, q = tid & 3;
      float p1 = 0.f;
      for (int t5 = q * 5; t5 < q * 5 + 5; ++t5) {
        const float* Mr = Mstu + ((size_t)t5 * DD + d) * NN;
        __builtin_prefetch(Mr + NN * DD, 0, 2);
#pragma unroll 8
        for (int n = 0; n < NN; ++n) p1 += Mr[n] * vu_s[n * 32 + t5];
      }
      float p2 = 0.f;
      const float* Ar = A + d * DD + q * 64;
      const float* xr = x_s + q * 64;
#pragma unroll 8
      for (int j = 0; j < 64; ++j) p2 += Ar[j] * xr[j];
      if (q == 0) {
        const float* Br = Bm + d * NN;
#pragma unroll 8
        for (int j = 0; j < NN; ++j) p2 += Br[j] * u_s[j];
      }
      p1s[tid] = p1;
      p2s[tid] = p2;
    }
    __syncthreads();
    if (tid < DD) {
      const float xn = p1s[4 * tid] + p1s[4 * tid + 1] + p1s[4 * tid + 2] + p1s[4 * tid + 3];
      const float ax = p2s[4 * tid] + p2s[4 * tid + 1] + p2s[4 * tid + 2] + p2s[4 * tid + 3];
      Wg[(size_t)tid * TT + headu] = (bf16)(xn - ax);   // roll + insert new_w
      x_s[tid] = xn;                                     // x <- x_next
    }
    __threadfence();   // make W column visible to next step's WMMA loads
    __syncthreads();
    head = headu;
  }
}

extern "C" void kernel_launch(void* const* d_in, const int* in_sizes, int n_in,
                              void* d_out, int out_size, void* d_ws, size_t ws_size,
                              hipStream_t stream) {
  (void)in_sizes; (void)n_in; (void)out_size; (void)ws_size;
  const float* A    = (const float*)d_in[0];
  const float* Bm   = (const float*)d_in[1];
  const float* Q    = (const float*)d_in[2];
  const float* R    = (const float*)d_in[3];
  const float* M    = (const float*)d_in[4];
  const float* Mstu = (const float*)d_in[5];
  const float* x0   = (const float*)d_in[6];
  const float* w0   = (const float*)d_in[7];
  const float* phi  = (const float*)d_in[8];
  const float* phs  = (const float*)d_in[9];
  float* out = (float*)d_out;

  char* ws = (char*)d_ws;                 // ws layout: M (1MB) | G (1MB) | W bf16 (256KB)
  float* Mw = (float*)(ws);
  float* Gw = (float*)(ws + (1u << 20));
  bf16*  Wg = (bf16*)(ws + (2u << 20));

  osfc_stu_kernel<<<dim3(1), dim3(1024), 0, stream>>>(
      A, Bm, Q, R, M, Mstu, x0, w0, phi, phs, out, Mw, Gw, Wg);
}
